// TransformerBlock_1331439861786
// MI455X (gfx1250) — matrix operand order repaired and checked by reading, not compiled, not run on hardware
//
#include <hip/hip_runtime.h>
#include <hip/hip_bf16.h>
#include <math.h>

// ---------------------------------------------------------------------------
// Types for CDNA5 WMMA (wave32): A/B = 16 bf16 per lane (8 VGPRs),
// C/D = 8 f32 per lane (8 VGPRs).
// ---------------------------------------------------------------------------
typedef __bf16 bf16;
typedef __attribute__((ext_vector_type(16))) __bf16 bf16x16;
typedef __attribute__((ext_vector_type(8)))  __bf16 bf16x8;
typedef __attribute__((ext_vector_type(8)))  float  f32x8;

#define DEVI static __device__ __forceinline__

// CDNA5 async global->LDS path (guarded so host pass / older toolchains fall
// back to plain load + ds_store; device pass on gfx1250 emits
// global_load_async_to_lds_b128 + s_wait_asynccnt).
// Builtin prototype (from compiler diagnostic): (v4i AS1*, v4i AS3*, Imm, Imm).
#if __has_builtin(__builtin_amdgcn_global_load_async_to_lds_b128) && \
    __has_builtin(__builtin_amdgcn_s_wait_asynccnt)
#define HAVE_ASYNC_LDS 1
typedef int v4i __attribute__((vector_size(16)));
typedef __attribute__((address_space(1))) v4i as1_v4i;
typedef __attribute__((address_space(3))) v4i as3_v4i;
#define WAIT_ASYNC(n) __builtin_amdgcn_s_wait_asynccnt(n)
#else
#define HAVE_ASYNC_LDS 0
#define WAIT_ASYNC(n)
#endif

DEVI f32x8 zero8() {
  f32x8 z;
#pragma unroll
  for (int i = 0; i < 8; ++i) z[i] = 0.0f;
  return z;
}

DEVI f32x8 wmma_bf16f32(bf16x16 a, bf16x16 b, f32x8 c) {
  // D = A(16x32) * B(32x16) + C(16x16), f32 accumulate
  return __builtin_amdgcn_wmma_f32_16x16x32_bf16(
      /*neg_a=*/false, a, /*neg_b=*/false, b,
      /*c_mod=*/(short)0, c, /*reuse_a=*/false, /*reuse_b=*/false);
}

// A-matrix fragment (16x32 bf16, row-major source, lda = row stride).
// Layout: lane half=lane/16, m=lane%16; VGPR0-7 hold contiguous
// K=half*16..+16 (2 bf16 per VGPR). Two contiguous 16B loads.
DEVI bf16x16 load_a_frag(const bf16* __restrict__ A, int lda, int m0, int k0, int lane) {
  const int m = m0 + (lane & 15);
  const int half = lane >> 4;
  const bf16* p = A + (size_t)m * lda + (size_t)(k0 + half * 16);
  bf16x8 lo = *(const bf16x8*)p;          // K = k0+half*16 .. +8
  bf16x8 hi = *(const bf16x8*)(p + 8);    // K = k0+half*16+8 .. +8
  bf16x16 r;
#pragma unroll
  for (int i = 0; i < 8; ++i) { r[i] = lo[i]; r[i + 8] = hi[i]; }
  return r;
}

// B-matrix fragment (32x16 bf16) where source W is [N][K] row-major and we
// compute A * W^T.  Layout used here (same contiguous k order as load_a_frag): lane holds column n=lane%16, contiguous
// K range of 16 starting at k0 + (lane/16)*16 -> one 32B contiguous load.
DEVI bf16x16 load_b_frag(const bf16* __restrict__ W, int ldw, int n0, int k0, int lane) {
  const int n = n0 + (lane & 15);
  const int ks = k0 + (lane >> 4) * 16;
  return *(const bf16x16*)(W + (size_t)n * ldw + (size_t)ks);
}

// Stage one 64(N) x 32(K) bf16 W tile into LDS (row-major [64][32]).
// 256 threads, 16B per thread, one async b128 issue each.
DEVI void stage_b_tile(const bf16* __restrict__ W, int ldw, int n0, int k0,
                       bf16* dst, int tid) {
  const int row = tid >> 2;   // 0..63 (N within tile)
  const int seg = tid & 3;    // 4 x 8 bf16 segments per 32-wide row
  const bf16* g = W + (size_t)(n0 + row) * ldw + (size_t)(k0 + seg * 8);
  bf16* l = dst + row * 32 + seg * 8;
#if HAVE_ASYNC_LDS
  __builtin_amdgcn_global_load_async_to_lds_b128((as1_v4i*)g, (as3_v4i*)l, 0, 0);
#else
  *(bf16x8*)l = *(const bf16x8*)g;
#endif
}

// ---------------------------------------------------------------------------
// f32 -> bf16 cast (grid-stride)
// ---------------------------------------------------------------------------
__global__ __launch_bounds__(256)
void cvt_bf16_kernel(const float* __restrict__ in, bf16* __restrict__ out, size_t n) {
  for (size_t i = (size_t)blockIdx.x * blockDim.x + threadIdx.x; i < n;
       i += (size_t)gridDim.x * blockDim.x)
    out[i] = (bf16)in[i];
}

// ---------------------------------------------------------------------------
// RMSNorm over C=2048, one block per token row; bf16 output for WMMA GEMMs.
// ---------------------------------------------------------------------------
__global__ __launch_bounds__(256)
void rmsnorm_bf16_kernel(const float* __restrict__ x, const float* __restrict__ w,
                         bf16* __restrict__ out, int C) {
  const float* row = x + (size_t)blockIdx.x * C;
  float ss = 0.0f;
  for (int i = threadIdx.x; i < C; i += 256) { float v = row[i]; ss += v * v; }
#pragma unroll
  for (int m = 1; m < 32; m <<= 1) ss += __shfl_xor(ss, m, 32);
  __shared__ float red[8];
  if ((threadIdx.x & 31) == 0) red[threadIdx.x >> 5] = ss;
  __syncthreads();
  float tot = 0.0f;
#pragma unroll
  for (int i = 0; i < 8; ++i) tot += red[i];
  const float inv = rsqrtf(tot / (float)C + 1e-6f);
  bf16* orow = out + (size_t)blockIdx.x * C;
  for (int i = threadIdx.x; i < C; i += 256)
    orow[i] = (bf16)(row[i] * inv * w[i]);
}

// ---------------------------------------------------------------------------
// Generic WMMA GEMM: out[M,N] = A[M,K](bf16) * W[N,K]^T(bf16) (+bias)(+resid)
// 256 threads = 8 waves; block tile 128(M) x 64(N); wave tile 16 x 64.
// The shared 32x64 B tile is staged into LDS with the async global->LDS path,
// double-buffered one K-step ahead (stage -> s_wait_asynccnt(1) -> barrier).
// Last K-iteration is peeled so the steady-state body is branch-free.
// ---------------------------------------------------------------------------
__global__ __launch_bounds__(256)
void gemm_bf16_kernel(const bf16* __restrict__ A, const bf16* __restrict__ W,
                      const float* __restrict__ bias, const float* __restrict__ resid,
                      float* __restrict__ outF, bf16* __restrict__ outB,
                      int M, int N, int K) {
  __shared__ __align__(32) bf16 btile[2][64 * 32];  // 2 x 4KB double buffer
  const int tid = threadIdx.x;
  const int lane = tid & 31;
  const int wave = tid >> 5;
  const int half = lane >> 4, lm = lane & 15;
  const int m0 = blockIdx.y * 128 + wave * 16;
  const int n0 = blockIdx.x * 64;

  f32x8 acc[4];
#pragma unroll
  for (int f = 0; f < 4; ++f) acc[f] = zero8();

  // compute step: issue all 8 ds_load_b128 into distinct values first, then
  // run the 4 WMMAs so dscnt waits can drain progressively.
  auto kstep = [&](int k0, const bf16* bt) {
    __builtin_prefetch(A + (size_t)(m0 + lm) * K + (size_t)(k0 + 32), 0, 3);
    bf16x16 a  = load_a_frag(A, K, m0, k0, lane);
    const bf16* p0 = bt + (size_t)((0 * 16 + lm) * 32 + half * 16);
    const bf16* p1 = bt + (size_t)((1 * 16 + lm) * 32 + half * 16);
    const bf16* p2 = bt + (size_t)((2 * 16 + lm) * 32 + half * 16);
    const bf16* p3 = bt + (size_t)((3 * 16 + lm) * 32 + half * 16);
    bf16x16 b0 = *(const bf16x16*)p0;
    bf16x16 b1 = *(const bf16x16*)p1;
    bf16x16 b2 = *(const bf16x16*)p2;
    bf16x16 b3 = *(const bf16x16*)p3;
    acc[0] = wmma_bf16f32(a, b0, acc[0]);
    acc[1] = wmma_bf16f32(a, b1, acc[1]);
    acc[2] = wmma_bf16f32(a, b2, acc[2]);
    acc[3] = wmma_bf16f32(a, b3, acc[3]);
  };

  stage_b_tile(W, K, n0, 0, &btile[0][0], tid);

  const int kiters = K / 32;  // always >= 2 here
  for (int i = 0; i < kiters - 1; ++i) {
    stage_b_tile(W, K, n0, (i + 1) * 32, &btile[(i + 1) & 1][0], tid);
    WAIT_ASYNC(1);   // previous stage complete; next stays in flight
    __syncthreads();
    kstep(i * 32, &btile[i & 1][0]);
    __syncthreads();  // reads done before next stage overwrites this buffer
  }
  WAIT_ASYNC(0);
  __syncthreads();
  kstep((kiters - 1) * 32, &btile[(kiters - 1) & 1][0]);

#pragma unroll
  for (int f = 0; f < 4; ++f) {
    const int col = n0 + f * 16 + lm;
    const float bv = bias ? bias[col] : 0.0f;
#pragma unroll
    for (int r = 0; r < 8; ++r) {
      const int row = m0 + r + 8 * half;
      const size_t idx = (size_t)row * N + col;
      float v = acc[f][r] + bv;
      if (resid) v += resid[idx];
      if (outF) outF[idx] = v;
      else      outB[idx] = (bf16)v;
    }
  }
}

// ---------------------------------------------------------------------------
// PoPE: lin[B*T, NH*128] -> out[B, NH, T, 256] bf16
// ---------------------------------------------------------------------------
__global__ __launch_bounds__(256)
void pope_kernel(const float* __restrict__ lin, const float* __restrict__ cosT,
                 const float* __restrict__ sinT, bf16* __restrict__ outp,
                 int Bn, int T, int NH) {
  const size_t idx = (size_t)blockIdx.x * blockDim.x + threadIdx.x;
  const size_t total = (size_t)Bn * T * NH * 128;
  if (idx >= total) return;
  const int i = (int)(idx & 127);
  size_t rest = idx >> 7;
  const int h = (int)(rest % NH); rest /= NH;
  const int t = (int)(rest % T);
  const int b = (int)(rest / T);
  const float x = lin[(size_t)(b * T + t) * (size_t)(NH * 128) + h * 128 + i];
  const float sp = (x > 20.0f) ? x : log1pf(__expf(x));
  const float c = cosT[t * 128 + i];
  const float s = sinT[t * 128 + i];
  const size_t o = (((size_t)(b * NH + h) * T + t) << 8) + (size_t)(2 * i);
  outp[o]     = (bf16)(sp * c);
  outp[o + 1] = (bf16)(sp * s);
}

// ---------------------------------------------------------------------------
// V transpose: vin[B*T, KVH*128] f32 -> vt[B, KVH, 128, T] bf16
// ---------------------------------------------------------------------------
__global__ __launch_bounds__(256)
void vtrans_kernel(const float* __restrict__ vin, bf16* __restrict__ vt,
                   int Bn, int T, int KVHn) {
  const size_t idx = (size_t)blockIdx.x * blockDim.x + threadIdx.x;
  const size_t total = (size_t)Bn * KVHn * 128 * T;
  if (idx >= total) return;
  const int t = (int)(idx % T);
  size_t rest = idx / T;
  const int d = (int)(rest % 128); rest /= 128;
  const int kh = (int)(rest % KVHn);
  const int b = (int)(rest / KVHn);
  vt[idx] = (bf16)vin[(size_t)(b * T + t) * (size_t)(KVHn * 128) + kh * 128 + d];
}

// ---------------------------------------------------------------------------
// Flash attention (causal), WMMA bf16, software-pipelined fragment loads.
// grid = (T/64, B*H); block = 128 threads (4 waves); wave = 16 query rows.
// qp: [B,H,T,256]  kp: [B,KVH,T,256]  vt: [B,KVH,128,T]  attn: [B*T, H*128]
// ---------------------------------------------------------------------------
__global__ __launch_bounds__(128)
void flash_attn_kernel(const bf16* __restrict__ qp, const bf16* __restrict__ kp,
                       const bf16* __restrict__ vt, bf16* __restrict__ attn,
                       int Bn, int Hn, int KVHn, int T) {
  const int lane = threadIdx.x & 31;
  const int wave = threadIdx.x >> 5;
  const int half = lane >> 4, lm = lane & 15;
  const int bh = blockIdx.y;
  const int b = bh / Hn, h = bh % Hn;
  const int kh = h / (Hn / KVHn);
  const int qt = blockIdx.x;
  const int qr0 = qt * 64 + wave * 16;

  const bf16* Q  = qp + (size_t)bh * T * 256;
  const bf16* Kp = kp + (size_t)(b * KVHn + kh) * T * 256;
  const bf16* Vt = vt + (size_t)(b * KVHn + kh) * 128 * (size_t)T;

  __shared__ __align__(32) bf16 plds[4][16 * 32];  // per-wave P staging

  bf16x16 qf[8];
#pragma unroll
  for (int c = 0; c < 8; ++c) qf[c] = load_a_frag(Q, 256, qr0, c * 32, lane);

  f32x8 o[8];
#pragma unroll
  for (int j = 0; j < 8; ++j) o[j] = zero8();
  float mrow[8], lrow[8];
#pragma unroll
  for (int r = 0; r < 8; ++r) { mrow[r] = -1e30f; lrow[r] = 0.0f; }

  const int ktiles = (qt + 1) * 2;  // uniform across waves -> safe barriers
  for (int kt = 0; kt < ktiles; ++kt) {
    const int k0 = kt * 32;
    // S(16x32) = Q(16x256) @ K^T(256x32), pipelined: preload chunk c+1
    f32x8 s0 = zero8(), s1 = zero8();
    bf16x16 kb0 = load_b_frag(Kp, 256, k0,      0, lane);
    bf16x16 kb1 = load_b_frag(Kp, 256, k0 + 16, 0, lane);
#pragma unroll
    for (int c = 0; c < 8; ++c) {
      const int kn = (c < 7) ? (c + 1) * 32 : 7 * 32;
      bf16x16 nb0 = load_b_frag(Kp, 256, k0,      kn, lane);
      bf16x16 nb1 = load_b_frag(Kp, 256, k0 + 16, kn, lane);
      s0 = wmma_bf16f32(qf[c], kb0, s0);
      s1 = wmma_bf16f32(qf[c], kb1, s1);
      kb0 = nb0; kb1 = nb1;
    }
    // online softmax (row = qr0 + r + 8*half ; cols = k0+lm, k0+16+lm)
    const int col0 = k0 + lm, col1 = k0 + 16 + lm;
    float p0v[8], p1v[8];
#pragma unroll
    for (int r = 0; r < 8; ++r) {
      const int row = qr0 + r + 8 * half;
      float v0 = (col0 <= row) ? s0[r] * 0.0625f : -3.0e38f;
      float v1 = (col1 <= row) ? s1[r] * 0.0625f : -3.0e38f;
      float t = fmaxf(v0, v1);
#pragma unroll
      for (int mm = 1; mm < 16; mm <<= 1) t = fmaxf(t, __shfl_xor(t, mm, 32));
      const float mn = fmaxf(mrow[r], t);
      const float alpha = __expf(mrow[r] - mn);
      mrow[r] = mn;
      const float p0 = __expf(v0 - mn);
      const float p1 = __expf(v1 - mn);
      float rs = p0 + p1;
#pragma unroll
      for (int mm = 1; mm < 16; mm <<= 1) rs += __shfl_xor(rs, mm, 32);
      lrow[r] = lrow[r] * alpha + rs;
#pragma unroll
      for (int j = 0; j < 8; ++j) o[j][r] *= alpha;
      p0v[r] = p0; p1v[r] = p1;
    }
    // C-layout -> A-layout for P via LDS
    bf16* pl = &plds[wave][0];
#pragma unroll
    for (int r = 0; r < 8; ++r) {
      const int row = r + 8 * half;
      pl[row * 32 + lm]      = (bf16)p0v[r];
      pl[row * 32 + 16 + lm] = (bf16)p1v[r];
    }
    __syncthreads();
    bf16x16 pf;
    {
      const bf16* pp = pl + lm * 32 + half * 16;
      bf16x8 lo = *(const bf16x8*)pp;
      bf16x8 hi = *(const bf16x8*)(pp + 8);
#pragma unroll
      for (int i = 0; i < 8; ++i) { pf[i] = lo[i]; pf[i + 8] = hi[i]; }
    }
    __syncthreads();
    // O(16x128) += P(16x32) @ V(32x128), pipelined V fragments
    bf16x16 vb = load_b_frag(Vt, T, 0, k0, lane);
#pragma unroll
    for (int j = 0; j < 8; ++j) {
      bf16x16 nvb = load_b_frag(Vt, T, ((j + 1) & 7) * 16, k0, lane);
      o[j] = wmma_bf16f32(pf, vb, o[j]);
      vb = nvb;
    }
  }

  // epilogue: attn[(b*T+row), h*128 + d] = O/l   (bf16 for the O-proj GEMM)
#pragma unroll
  for (int j = 0; j < 8; ++j) {
#pragma unroll
    for (int r = 0; r < 8; ++r) {
      const int row = qr0 + r + 8 * half;
      const float v = o[j][r] / lrow[r];
      attn[(size_t)(b * T + row) * 2048 + (size_t)(h * 128 + j * 16 + lm)] = (bf16)v;
    }
  }
}

// ---------------------------------------------------------------------------
// SwiGLU combine: ffn = silu(g) * u  (bf16 in/out)
// ---------------------------------------------------------------------------
__global__ __launch_bounds__(256)
void silu_mul_kernel(const bf16* __restrict__ g, const bf16* __restrict__ u,
                     bf16* __restrict__ o, size_t n) {
  for (size_t i = (size_t)blockIdx.x * blockDim.x + threadIdx.x; i < n;
       i += (size_t)gridDim.x * blockDim.x) {
    const float x = (float)g[i];
    const float s = x / (1.0f + __expf(-x));
    o[i] = (bf16)(s * (float)u[i]);
  }
}

// ---------------------------------------------------------------------------
// Host orchestration
// ---------------------------------------------------------------------------
extern "C" void kernel_launch(void* const* d_in, const int* in_sizes, int n_in,
                              void* d_out, int out_size, void* d_ws, size_t ws_size,
                              hipStream_t stream) {
  (void)in_sizes; (void)n_in; (void)out_size; (void)ws_size;
  const int B = 2, T = 2048, C = 2048, H = 16, KVH = 4, HIDDEN = 5504;
  const int M = B * T;            // 4096 token rows
  const int KVD = KVH * 128;      // 512

  const float* x    = (const float*)d_in[0];
  const float* cosT = (const float*)d_in[2];
  const float* sinT = (const float*)d_in[3];
  const float* ln1w = (const float*)d_in[4];
  const float* qwF  = (const float*)d_in[5];
  const float* qb   = (const float*)d_in[6];
  const float* kwF  = (const float*)d_in[7];
  const float* kb   = (const float*)d_in[8];
  const float* vwF  = (const float*)d_in[9];
  const float* vb   = (const float*)d_in[10];
  const float* owF  = (const float*)d_in[11];
  const float* ob   = (const float*)d_in[12];
  const float* ln2w = (const float*)d_in[13];
  const float* wgF  = (const float*)d_in[14];
  const float* w1F  = (const float*)d_in[15];
  const float* w2F  = (const float*)d_in[16];
  float* outF = (float*)d_out;

  char* ws = (char*)d_ws;
  size_t off = 0;
  auto alloc = [&](size_t bytes) -> void* {
    void* p = ws + off;
    off = (off + bytes + 255) & ~(size_t)255;
    return p;
  };

  bf16* wq = (bf16*)alloc((size_t)C * C * 2);
  bf16* wk = (bf16*)alloc((size_t)KVD * C * 2);
  bf16* wv = (bf16*)alloc((size_t)KVD * C * 2);
  bf16* wo = (bf16*)alloc((size_t)C * C * 2);
  bf16* wg = (bf16*)alloc((size_t)HIDDEN * C * 2);
  bf16* w1 = (bf16*)alloc((size_t)HIDDEN * C * 2);
  bf16* w2 = (bf16*)alloc((size_t)C * HIDDEN * 2);
  bf16* h1   = (bf16*)alloc((size_t)M * C * 2);
  float* qo  = (float*)alloc((size_t)M * C * 4);
  float* ko  = (float*)alloc((size_t)M * KVD * 4);
  float* vo  = (float*)alloc((size_t)M * KVD * 4);
  bf16* qpo  = (bf16*)alloc((size_t)B * H * T * 256 * 2);
  bf16* kpo  = (bf16*)alloc((size_t)B * KVH * T * 256 * 2);
  bf16* vtb  = (bf16*)alloc((size_t)B * KVH * 128 * T * 2);
  bf16* attn = (bf16*)alloc((size_t)M * C * 2);
  bf16* h2   = (bf16*)alloc((size_t)M * C * 2);
  bf16* gbuf = (bf16*)alloc((size_t)M * HIDDEN * 2);
  bf16* ubuf = (bf16*)alloc((size_t)M * HIDDEN * 2);
  bf16* ffn  = (bf16*)alloc((size_t)M * HIDDEN * 2);

  auto cvt = [&](const float* src, bf16* dst, size_t n) {
    cvt_bf16_kernel<<<dim3(2048), dim3(256), 0, stream>>>(src, dst, n);
  };
  auto gemm = [&](const bf16* A, const bf16* W, const float* bias, const float* resid,
                  float* oF, bf16* oB, int m, int n, int k) {
    gemm_bf16_kernel<<<dim3(n / 64, m / 128), dim3(256), 0, stream>>>(
        A, W, bias, resid, oF, oB, m, n, k);
  };

  // 0) weights -> bf16 (L2-resident working set)
  cvt(qwF, wq, (size_t)C * C);
  cvt(kwF, wk, (size_t)KVD * C);
  cvt(vwF, wv, (size_t)KVD * C);
  cvt(owF, wo, (size_t)C * C);
  cvt(wgF, wg, (size_t)HIDDEN * C);
  cvt(w1F, w1, (size_t)HIDDEN * C);
  cvt(w2F, w2, (size_t)C * HIDDEN);

  // 1) RMSNorm(ln1) -> h1 (bf16)
  rmsnorm_bf16_kernel<<<dim3(M), dim3(256), 0, stream>>>(x, ln1w, h1, C);

  // 2) Q,K,V projections (WMMA GEMMs, bias folded in)
  gemm(h1, wq, qb, nullptr, qo, nullptr, M, C, C);
  gemm(h1, wk, kb, nullptr, ko, nullptr, M, KVD, C);
  gemm(h1, wv, vb, nullptr, vo, nullptr, M, KVD, C);

  // 3) PoPE on Q and K; transpose V
  {
    const size_t nq = (size_t)B * T * H * 128;
    pope_kernel<<<dim3((unsigned)((nq + 255) / 256)), dim3(256), 0, stream>>>(
        qo, cosT, sinT, qpo, B, T, H);
    const size_t nk = (size_t)B * T * KVH * 128;
    pope_kernel<<<dim3((unsigned)((nk + 255) / 256)), dim3(256), 0, stream>>>(
        ko, cosT, sinT, kpo, B, T, KVH);
    const size_t nv = (size_t)B * KVH * 128 * T;
    vtrans_kernel<<<dim3((unsigned)((nv + 255) / 256)), dim3(256), 0, stream>>>(
        vo, vtb, B, T, KVH);
  }

  // 4) causal flash attention (WMMA)
  flash_attn_kernel<<<dim3(T / 64, B * H), dim3(128), 0, stream>>>(
      qpo, kpo, vtb, attn, B, H, KVH, T);

  // 5) O-projection + residual -> x1 (stored in d_out)
  gemm(attn, wo, ob, x, outF, nullptr, M, C, C);

  // 6) RMSNorm(ln2) -> h2
  rmsnorm_bf16_kernel<<<dim3(M), dim3(256), 0, stream>>>(outF, ln2w, h2, C);

  // 7) FFN: gate/up GEMMs -> SwiGLU -> down GEMM + residual into d_out
  gemm(h2, wg, nullptr, nullptr, nullptr, gbuf, M, HIDDEN, C);
  gemm(h2, w1, nullptr, nullptr, nullptr, ubuf, M, HIDDEN, C);
  silu_mul_kernel<<<dim3(4096), dim3(256), 0, stream>>>(
      gbuf, ubuf, ffn, (size_t)M * HIDDEN);
  gemm(ffn, w2, nullptr, outF, outF, nullptr, M, C, HIDDEN);
}
